// RGCNLayer_78391743086960
// MI455X (gfx1250) — matrix operand order repaired and checked by reading, not compiled, not run on hardware
//
#include <hip/hip_runtime.h>
#include <hip/hip_bf16.h>

// ---- CDNA5 WMMA types ----
typedef __attribute__((ext_vector_type(16))) __bf16 v16bf;
typedef __attribute__((ext_vector_type(8)))  float  v8f;

#define HD      256
#define NREL    16
#define NNODES  50000
#define NEDGES  800000
#define MTILE   64                       // edges (rows) per block
#define MAXTILES (NEDGES / MTILE + NREL) // worst-case #64-edge tiles after per-relation split

struct Frag32B { uint4 lo; uint4 hi; };

// A fragment (16x32 bf16, MxK): lane m = l&15, half = l>>4.
// VGPR j holds K = k0 + half*16 + {2j, 2j+1} (same order as B).
// -> 32 contiguous bytes in a row-major bf16 row.
__device__ __forceinline__ v16bf load_a_frag(const __bf16* p) {
  Frag32B f;
  f.lo = *(const uint4*)(p);
  f.hi = *(const uint4*)(p + 8);
  return __builtin_bit_cast(v16bf, f);
}

// B fragment (32x16 bf16, KxN): lane n = l&15, half = l>>4.
// VGPR j holds K = k0 + half*16 + {2j, 2j+1}. With W pre-transposed to Wt[n][k]
// (column-major of W), this is 32 contiguous bytes.
__device__ __forceinline__ v16bf load_b_frag(const __bf16* p) {
  Frag32B f;
  f.lo = *(const uint4*)(p);
  f.hi = *(const uint4*)(p + 8);
  return __builtin_bit_cast(v16bf, f);
}

// ---------------- precision conversion / packing ----------------
struct B4 { __bf16 a, b, c, d; };

__global__ void k_cvt_x(const float* __restrict__ x, __bf16* __restrict__ xbf, int n4) {
  int i = blockIdx.x * blockDim.x + threadIdx.x;
  if (i >= n4) return;
  float4 v = ((const float4*)x)[i];
  B4 o = { (__bf16)v.x, (__bf16)v.y, (__bf16)v.z, (__bf16)v.w };
  ((uint2*)xbf)[i] = __builtin_bit_cast(uint2, o);
}

// Wt[r][n][k] = W_rel[r][k][n] (bf16); slot r==NREL handles W_self.
__global__ void k_transpose_w(const float* __restrict__ Wrel, const float* __restrict__ Wself,
                              __bf16* __restrict__ wtRel, __bf16* __restrict__ wtSelf) {
  int i = blockIdx.x * blockDim.x + threadIdx.x;      // over (NREL+1)*HD*HD
  int r   = i >> 16;
  int rem = i & 65535;
  int n = rem >> 8;
  int k = rem & 255;
  const float* W = (r < NREL) ? (Wrel + r * HD * HD) : Wself;
  __bf16*      T = (r < NREL) ? (wtRel + r * HD * HD) : wtSelf;
  T[n * HD + k] = (__bf16)W[k * HD + n];
}

// ---------------- edge binning by relation ----------------
__global__ void k_zero(int* counts) {
  if (threadIdx.x < NREL) counts[threadIdx.x] = 0;
}

__global__ void k_hist(const int* __restrict__ et, int* __restrict__ counts, int E) {
  int e = blockIdx.x * blockDim.x + threadIdx.x;
  if (e < E) atomicAdd(&counts[et[e]], 1);
}

__global__ void k_scan(const int* __restrict__ counts, int* offsets, int* cursors, int* tileOff) {
  if (threadIdx.x != 0) return;
  int o = 0, t = 0;
  for (int r = 0; r < NREL; ++r) {
    offsets[r] = o; cursors[r] = o; tileOff[r] = t;
    int c = counts[r];
    o += c;
    t += (c + MTILE - 1) >> 6;
  }
  offsets[NREL] = o;
  tileOff[NREL] = t;
}

__global__ void k_scatter(const int* __restrict__ et, int* __restrict__ cursors,
                          int* __restrict__ perm, int E) {
  int e = blockIdx.x * blockDim.x + threadIdx.x;
  if (e < E) {
    int p = atomicAdd(&cursors[et[e]], 1);
    perm[p] = e;
  }
}

// ---------------- self-loop GEMM: out = x @ W_self + bias ----------------
// 16 waves/block in a 4x4 grid: wave = (mi, ni); block = 64 rows x 256 cols.
// Each wave: 16 rows x 64 cols (4 N-tiles) -> A frag loaded once per K-step,
// reused by 4 WMMAs; B frags shared by the 4 waves of the same ni via WGP$.
__global__ __launch_bounds__(512) void k_self_gemm(const __bf16* __restrict__ xbf,
                                                   const __bf16* __restrict__ wtSelf,
                                                   const float* __restrict__ bias,
                                                   float* __restrict__ out) {
  int wave = threadIdx.x >> 5;
  int mi   = wave >> 2;
  int ni   = wave & 3;
  int m0   = blockIdx.x * MTILE + mi * 16;
  if (m0 >= NNODES) return;            // uniform per wave; no barriers follow

  int lane = threadIdx.x & 31;
  int ln   = lane & 15;
  int half = lane >> 4;
  int n0   = ni * 64;

  const __bf16* pA = xbf + (m0 + ln) * HD + half * 16;    // A row (M = ln)
  const __bf16* pB = wtSelf + (n0 + ln) * HD + half * 16; // B cols start (N = n0+ln)

  v8f acc[4];
#pragma unroll
  for (int t = 0; t < 4; ++t) acc[t] = v8f{0.f, 0.f, 0.f, 0.f, 0.f, 0.f, 0.f, 0.f};

#pragma unroll
  for (int k0 = 0; k0 < HD; k0 += 32) {
    v16bf a = load_a_frag(pA + k0);
#pragma unroll
    for (int t = 0; t < 4; ++t) {
      v16bf b = load_b_frag(pB + (size_t)(t * 16) * HD + k0);  // N-tile t: +16 Wt rows
      acc[t] = __builtin_amdgcn_wmma_f32_16x16x32_bf16(false, a, false, b,
                                                       (short)0, acc[t], false, false);
    }
  }

#pragma unroll
  for (int t = 0; t < 4; ++t) {
    int n = n0 + t * 16 + ln;
    float bi = bias[n];
#pragma unroll
    for (int v = 0; v < 8; ++v) {
      int m = m0 + half * 8 + v;       // C layout: M = v + 8*half, N = ln
      out[m * HD + n] = acc[t][v] + bi;
    }
  }
}

// ---------------- relation-binned edge GEMM + atomic scatter ----------------
// One block per 64-edge tile of one relation; 16 waves in a 4x4 (mi, ni) grid.
__global__ __launch_bounds__(512) void k_edge_gemm(const __bf16* __restrict__ xbf,
                                                   const __bf16* __restrict__ wtRel,
                                                   const int* __restrict__ ei,      // [2, E]
                                                   const int* __restrict__ perm,
                                                   const int* __restrict__ offsets, // [NREL+1]
                                                   const int* __restrict__ tileOff, // [NREL+1]
                                                   float* __restrict__ out) {
  __shared__ int sSrc[MTILE];
  __shared__ int sDst[MTILE];

  int b = blockIdx.x;
  if (b >= tileOff[NREL]) return;     // uniform: static grid is worst-case sized

  int r = 0;
#pragma unroll
  for (int i = 1; i < NREL; ++i)
    if (b >= tileOff[i]) r = i;
  int base = offsets[r] + (b - tileOff[r]) * MTILE;
  int cnt  = min(MTILE, offsets[r + 1] - base);

  int tid = threadIdx.x;
  if (tid < MTILE) {
    int srcRow = 0, dstRow = -1;
    if (tid < cnt) {
      int e  = perm[base + tid];
      srcRow = ei[e];            // edge_index[0][e]
      dstRow = ei[NEDGES + e];   // edge_index[1][e]
    }
    sSrc[tid] = srcRow;
    sDst[tid] = dstRow;
  }
  __syncthreads();

  int wave = tid >> 5;
  int mi   = wave >> 2;
  int ni   = wave & 3;
  int lane = tid & 31;
  int ln   = lane & 15;
  int half = lane >> 4;
  int n0   = ni * 64;
  int mrow = mi * 16;

  const __bf16* pA = xbf + sSrc[mrow + ln] * HD + half * 16;         // gathered src row
  const __bf16* pB = wtRel + r * HD * HD + (n0 + ln) * HD + half * 16;

  v8f acc[4];
#pragma unroll
  for (int t = 0; t < 4; ++t) acc[t] = v8f{0.f, 0.f, 0.f, 0.f, 0.f, 0.f, 0.f, 0.f};

#pragma unroll
  for (int k0 = 0; k0 < HD; k0 += 32) {
    v16bf a = load_a_frag(pA + k0);                 // loaded once, reused by 4 WMMAs
#pragma unroll
    for (int t = 0; t < 4; ++t) {
      v16bf b = load_b_frag(pB + (size_t)(t * 16) * HD + k0);
      acc[t] = __builtin_amdgcn_wmma_f32_16x16x32_bf16(false, a, false, b,
                                                       (short)0, acc[t], false, false);
    }
  }

#pragma unroll
  for (int t = 0; t < 4; ++t) {
    int n = n0 + t * 16 + ln;
#pragma unroll
    for (int v = 0; v < 8; ++v) {
      int d = sDst[mrow + half * 8 + v];
      if (d >= 0) atomicAdd(&out[d * HD + n], acc[t][v]);  // L2-resident f32 atomics
    }
  }
}

__global__ void k_relu(float* __restrict__ out, int n) {
  int i = blockIdx.x * blockDim.x + threadIdx.x;
  if (i < n) out[i] = fmaxf(out[i], 0.f);
}

// ---------------- host launcher ----------------
extern "C" void kernel_launch(void* const* d_in, const int* in_sizes, int n_in,
                              void* d_out, int out_size, void* d_ws, size_t ws_size,
                              hipStream_t stream) {
  const float* x     = (const float*)d_in[0];
  const int*   ei    = (const int*)d_in[1];   // [2, E] (integer inputs arrive as int32)
  const int*   et    = (const int*)d_in[2];   // [E]
  const float* Wrel  = (const float*)d_in[3]; // [R, d, d]
  const float* Wself = (const float*)d_in[4]; // [d, d]
  const float* bias  = (const float*)d_in[5]; // [d]
  float* out = (float*)d_out;

  // scratch carve-out (~31 MB total)
  char* ws = (char*)d_ws;
  size_t off = 0;
  auto carve = [&](size_t bytes) -> void* {
    void* p = ws + off;
    off = (off + bytes + 255) & ~(size_t)255;
    return p;
  };
  __bf16* xbf    = (__bf16*)carve((size_t)NNODES * HD * 2);
  __bf16* wtRel  = (__bf16*)carve((size_t)NREL * HD * HD * 2);
  __bf16* wtSelf = (__bf16*)carve((size_t)HD * HD * 2);
  int* counts  = (int*)carve(NREL * 4);
  int* offsets = (int*)carve((NREL + 1) * 4);
  int* cursors = (int*)carve(NREL * 4);
  int* tileOff = (int*)carve((NREL + 1) * 4);
  int* perm    = (int*)carve((size_t)NEDGES * 4);

  // 1) precision conversion & weight packing for WMMA-friendly loads
  int n4 = NNODES * HD / 4;
  k_cvt_x<<<(n4 + 255) / 256, 256, 0, stream>>>(x, xbf, n4);
  k_transpose_w<<<((NREL + 1) * HD * HD) / 256, 256, 0, stream>>>(Wrel, Wself, wtRel, wtSelf);

  // 2) bin edges by relation
  k_zero<<<1, 32, 0, stream>>>(counts);
  k_hist<<<(NEDGES + 255) / 256, 256, 0, stream>>>(et, counts, NEDGES);
  k_scan<<<1, 32, 0, stream>>>(counts, offsets, cursors, tileOff);
  k_scatter<<<(NEDGES + 255) / 256, 256, 0, stream>>>(et, cursors, perm, NEDGES);

  // 3) self-loop GEMM writes out = x @ W_self + bias  (64-row blocks, uniform tail guard)
  int selfBlocks = (NNODES + MTILE - 1) / MTILE;
  k_self_gemm<<<selfBlocks, 512, 0, stream>>>(xbf, wtSelf, bias, out);

  // 4) relation-binned edge GEMM, atomic scatter-add into out
  k_edge_gemm<<<MAXTILES, 512, 0, stream>>>(xbf, wtRel, ei, perm, offsets, tileOff, out);

  // 5) ReLU
  k_relu<<<(NNODES * HD + 255) / 256, 256, 0, stream>>>(out, NNODES * HD);
}
